// EpipolarAttention_42752104464489
// MI455X (gfx1250) — matrix operand order repaired and checked by reading, not compiled, not run on hardware
//
#include <hip/hip_runtime.h>
#include <hip/hip_bf16.h>

#define Bb 8
#define Cc 1152
#define Hh 32
#define Ww 32
#define Nn 1024            // H*W
#define NBLK_N (Cc / 32)   // 36 n-blocks of 32 columns

typedef __attribute__((ext_vector_type(16))) _Float16 v16h;
typedef __attribute__((ext_vector_type(8)))  float    v8f;
typedef __attribute__((ext_vector_type(4)))  unsigned int v4u;

// ---------------- helpers ----------------

__device__ __forceinline__ v16h load_16h_2x(const _Float16* p0, const _Float16* p1) {
    union { v16h h; v4u u[2]; } r;
    r.u[0] = *(const v4u*)p0;   // global_load_b128
    r.u[1] = *(const v4u*)p1;   // global_load_b128
    return r.h;
}

__device__ __forceinline__ float wave_max(float v) {
    #pragma unroll
    for (int o = 16; o > 0; o >>= 1) v = fmaxf(v, __shfl_xor(v, o, 32));
    return v;
}
__device__ __forceinline__ float wave_sum(float v) {
    #pragma unroll
    for (int o = 16; o > 0; o >>= 1) v += __shfl_xor(v, o, 32);
    return v;
}

// ---------------- 1) epipolar line coefficients per (b, j) ----------------
// F = K2^-T * (skew(t) * R) * K1^-1  (SVD rank-2 projection is an exact identity:
// skew(t)R already has singular values (|t|,|t|,0)).
// line_j = F * (xj, yj, 1); a=l0/l2, b=l1/l2; d(i,j) = |a*xi + b*yi + 1| / sqrt(a^2+b^2)

__global__ void __launch_bounds__(256) epi_coef(const float* __restrict__ K1,
                                                const float* __restrict__ K2,
                                                const float* __restrict__ R,
                                                const float* __restrict__ t,
                                                float4* __restrict__ coef) {
    int gid = blockIdx.x * 256 + threadIdx.x;          // B*N threads
    int b = gid >> 10;
    int j = gid & (Nn - 1);
    const float* k1 = K1 + b * 9;
    const float* k2 = K2 + b * 9;
    const float* Rb = R  + b * 9;
    const float* tb = t  + b * 3;
    float t0 = tb[0], t1 = tb[1], t2 = tb[2];

    float E[3][3];
    #pragma unroll
    for (int c = 0; c < 3; ++c) {
        E[0][c] = -t2 * Rb[3 + c] + t1 * Rb[6 + c];
        E[1][c] =  t2 * Rb[0 + c] - t0 * Rb[6 + c];
        E[2][c] = -t1 * Rb[0 + c] + t0 * Rb[3 + c];
    }
    float fx1 = k1[0], cx1 = k1[2], fy1 = k1[4], cy1 = k1[5];
    float M[3][3];
    #pragma unroll
    for (int r = 0; r < 3; ++r) {
        M[r][0] = E[r][0] / fx1;
        M[r][1] = E[r][1] / fy1;
        M[r][2] = -E[r][0] * cx1 / fx1 - E[r][1] * cy1 / fy1 + E[r][2];
    }
    float fx2 = k2[0], cx2 = k2[2], fy2 = k2[4], cy2 = k2[5];
    float F0[3], F1[3], F2[3];
    #pragma unroll
    for (int c = 0; c < 3; ++c) {
        F0[c] = M[0][c] / fx2;
        F1[c] = M[1][c] / fy2;
        F2[c] = -cx2 / fx2 * M[0][c] - cy2 / fy2 * M[1][c] + M[2][c];
    }
    float xj = (float)(j >> 5);        // ix = j / W   (meshgrid 'ij')
    float yj = (float)(j & (Ww - 1));  // iy = j % W
    float l0 = F0[0] * xj + F0[1] * yj + F0[2];
    float l1 = F1[0] * xj + F1[1] * yj + F1[2];
    float l2 = F2[0] * xj + F2[1] * yj + F2[2];
    float a  = l0 / l2;
    float bl = l1 / l2;
    float inv = rsqrtf(a * a + bl * bl);
    coef[gid] = make_float4(a * inv, bl * inv, inv, 0.0f);
}

// ---------------- 2) f_src f32 -> f16 ----------------

__global__ void __launch_bounds__(256) epi_cvt(const float* __restrict__ src,
                                               _Float16* __restrict__ dst, int n) {
    int gid = blockIdx.x * 256 + threadIdx.x;
    if (gid < n) dst[gid] = (_Float16)src[gid];
}

// ---------------- 3) row softmax: w[i,j] = 1 - softmax_j(5*d) ----------------
// one wave per row i; coef reads and w writes are lane-coalesced

__global__ void __launch_bounds__(256) epi_rowsm(const float4* __restrict__ coef,
                                                 float* __restrict__ wbuf) {
    int wid  = threadIdx.x >> 5;
    int lane = threadIdx.x & 31;
    int row  = blockIdx.x * 8 + wid;        // 0 .. B*N-1
    int b = row >> 10;
    int i = row & (Nn - 1);
    float xi = (float)(i >> 5);
    float yi = (float)(i & (Ww - 1));
    const float4* cf = coef + b * Nn;

    float m = -1e30f;
    for (int j = lane; j < Nn; j += 32) {
        float4 c = cf[j];
        float d = fabsf(c.x * xi + c.y * yi + c.z);
        m = fmaxf(m, 5.0f * d);
    }
    m = wave_max(m);
    float s = 0.0f;
    for (int j = lane; j < Nn; j += 32) {
        float4 c = cf[j];
        float d = fabsf(c.x * xi + c.y * yi + c.z);
        s += __expf(5.0f * d - m);
    }
    s = wave_sum(s);
    float invs = 1.0f / s;
    float* wrow = wbuf + (size_t)row * Nn;
    for (int j = lane; j < Nn; j += 32) {
        float4 c = cf[j];
        float d = fabsf(c.x * xi + c.y * yi + c.z);
        wrow[j] = 1.0f - __expf(5.0f * d - m) * invs;
    }
}

// ---------------- 4) column softmax over i -> A (fp16, row-major i,j) ----------------
// 256-thread block handles 32 consecutive columns: thread = (row-group 0..7, col 0..31)
// -> every global access touches 32 consecutive elements (fully coalesced),
// cross-row-group max/sum reduced through an 8x32 LDS tile.

__global__ void __launch_bounds__(256) epi_colsm(const float* __restrict__ wbuf,
                                                 _Float16* __restrict__ A16) {
    __shared__ float red[8][32];

    const int c  = threadIdx.x & 31;        // column within group
    const int rg = threadIdx.x >> 5;        // row group 0..7
    const int jb = blockIdx.x * 32;         // first column of this block
    const int b  = blockIdx.y;

    const float* wb = wbuf + (size_t)b * Nn * Nn + jb + c;

    // pass 1: max over i
    float m = -1e30f;
    for (int i = rg; i < Nn; i += 8) m = fmaxf(m, wb[(size_t)i * Nn]);
    red[rg][c] = m;
    __syncthreads();
    #pragma unroll
    for (int o = 4; o > 0; o >>= 1) {
        if (rg < o) red[rg][c] = fmaxf(red[rg][c], red[rg + o][c]);
        __syncthreads();
    }
    m = red[0][c];
    __syncthreads();

    // pass 2: sum of exp(w - m) over i
    float s = 0.0f;
    for (int i = rg; i < Nn; i += 8) s += __expf(wb[(size_t)i * Nn] - m);
    red[rg][c] = s;
    __syncthreads();
    #pragma unroll
    for (int o = 4; o > 0; o >>= 1) {
        if (rg < o) red[rg][c] += red[rg + o][c];
        __syncthreads();
    }
    const float invs = 1.0f / red[0][c];

    // pass 3: emit A in fp16 (coalesced 64B rows)
    _Float16* ac = A16 + (size_t)b * Nn * Nn + jb + c;
    for (int i = rg; i < Nn; i += 8)
        ac[(size_t)i * Nn] = (_Float16)(__expf(wb[(size_t)i * Nn] - m) * invs);
}

// ---------------- 5) out[b,i,c] = sum_j A[b,i,j] * fsrc16[b,c,j]  (WMMA GEMM) ----------
// one wave per 32x32 output tile, 2x2 register-blocked 16x16 WMMA tiles, K step 32.
// A-tile 16x32 f16: lane m=lane&15, K-half = lane>>4; VGPR0-3 = K[half*8 .. +7],
// VGPR4-7 = K[16+half*8 .. +7]  -> two b128 loads.
// B-tile 32x16 f16: lane n=lane&15, K-half sel: VGPR0-3 = K[half*8 .. +7],
// VGPR4-7 = K[16+half*8 .. +7] -> two b128 loads (fsrc row c is contiguous in j=K).

__global__ void __launch_bounds__(256) epi_gemm(const _Float16* __restrict__ A16,
                                                const _Float16* __restrict__ F16,
                                                float* __restrict__ out) {
    const int wid  = threadIdx.x >> 5;
    const int lane = threadIdx.x & 31;
    const int b    = blockIdx.y;
    const int tile = blockIdx.x * 8 + wid;             // 0 .. 32*36-1
    const int mblk = tile / NBLK_N;
    const int nblk = tile - mblk * NBLK_N;
    const int m0 = mblk * 32;
    const int n0 = nblk * 32;

    const _Float16* Ab = A16 + (size_t)b * Nn * Nn;
    const _Float16* Fb = F16 + (size_t)b * Cc * Nn;

    const int r  = lane & 15;
    const int hf = lane >> 4;

    const _Float16* pa0 = Ab + (size_t)(m0 + r) * Nn + hf * 8;
    const _Float16* pa1 = pa0 + (size_t)16 * Nn;
    const _Float16* pb0 = Fb + (size_t)(n0 + r) * Nn + hf * 8;
    const _Float16* pb1 = pb0 + (size_t)16 * Nn;

    v8f acc00 = {}, acc01 = {}, acc10 = {}, acc11 = {};

    #pragma unroll 2
    for (int k = 0; k < Nn; k += 32) {
        v16h a0 = load_16h_2x(pa0, pa0 + 16);
        v16h a1 = load_16h_2x(pa1, pa1 + 16);
        v16h b0 = load_16h_2x(pb0, pb0 + 16);
        v16h b1 = load_16h_2x(pb1, pb1 + 16);
        __builtin_prefetch(pa0 + 32, 0, 3);            // global_prefetch_b8
        __builtin_prefetch(pb0 + 32, 0, 3);
        acc00 = __builtin_amdgcn_wmma_f32_16x16x32_f16(false, a0, false, b0, (short)0, acc00, false, false);
        acc01 = __builtin_amdgcn_wmma_f32_16x16x32_f16(false, a0, false, b1, (short)0, acc01, false, false);
        acc10 = __builtin_amdgcn_wmma_f32_16x16x32_f16(false, a1, false, b0, (short)0, acc10, false, false);
        acc11 = __builtin_amdgcn_wmma_f32_16x16x32_f16(false, a1, false, b1, (short)0, acc11, false, false);
        pa0 += 32; pa1 += 32; pb0 += 32; pb1 += 32;
    }

    // C/D layout: VGPR v -> row m = v + (lane>=16)*8, col = lane&15
    float* ob = out + (size_t)b * Nn * Cc;
    const int cn = lane & 15;
    const int mo = (lane >> 4) * 8;
    float* p00 = ob + (size_t)(m0 + mo) * Cc      + n0 + cn;
    float* p01 = p00 + 16;
    float* p10 = ob + (size_t)(m0 + 16 + mo) * Cc + n0 + cn;
    float* p11 = p10 + 16;
    #pragma unroll
    for (int v = 0; v < 8; ++v) {
        p00[(size_t)v * Cc] = acc00[v];
        p01[(size_t)v * Cc] = acc01[v];
        p10[(size_t)v * Cc] = acc10[v];
        p11[(size_t)v * Cc] = acc11[v];
    }
}

// ---------------- launch ----------------

extern "C" void kernel_launch(void* const* d_in, const int* in_sizes, int n_in,
                              void* d_out, int out_size, void* d_ws, size_t ws_size,
                              hipStream_t stream) {
    // inputs: 0 f_tar (unused by the math), 1 f_src, 2 K1, 3 K2, 4 R, 5 t
    const float* f_src = (const float*)d_in[1];
    const float* K1    = (const float*)d_in[2];
    const float* K2    = (const float*)d_in[3];
    const float* R     = (const float*)d_in[4];
    const float* t     = (const float*)d_in[5];
    float* out = (float*)d_out;

    char* ws = (char*)d_ws;
    const size_t OFF_COEF = 0;                                   // B*N float4  = 128 KB
    const size_t OFF_W    = OFF_COEF + (size_t)Bb * Nn * 16;     // B*N*N f32   = 33.6 MB
    const size_t OFF_A16  = OFF_W    + (size_t)Bb * Nn * Nn * 4; // B*N*N f16   = 16.8 MB
    const size_t OFF_F16  = OFF_A16  + (size_t)Bb * Nn * Nn * 2; // B*C*N f16   = 18.9 MB

    float4*   coef = (float4*)(ws + OFF_COEF);
    float*    wbuf = (float*)(ws + OFF_W);
    _Float16* A16  = (_Float16*)(ws + OFF_A16);
    _Float16* F16  = (_Float16*)(ws + OFF_F16);

    epi_coef<<<(Bb * Nn) / 256, 256, 0, stream>>>(K1, K2, R, t, coef);

    const int nsrc = Bb * Cc * Nn;
    epi_cvt<<<(nsrc + 255) / 256, 256, 0, stream>>>(f_src, F16, nsrc);

    epi_rowsm<<<(Bb * Nn) / 8, 256, 0, stream>>>(coef, wbuf);

    dim3 cgrid(Nn / 32, Bb);            // (32, 8), 32 columns per block
    epi_colsm<<<cgrid, 256, 0, stream>>>(wbuf, A16);

    dim3 grid((32 * NBLK_N) / 8, Bb);   // (144, 8)
    epi_gemm<<<grid, 256, 0, stream>>>(A16, F16, out);
}
